// ConvE_62758062129643
// MI455X (gfx1250) — matrix operand order repaired and checked by reading, not compiled, not run on hardware
//
#include <hip/hip_runtime.h>

typedef __attribute__((ext_vector_type(16))) _Float16 v16h;
typedef __attribute__((ext_vector_type(8)))  float    v8f;

#define EPS     1e-5f
#define Dm      128     // embedding dim / GEMM N
#define Cc      16      // conv channels
#define KTOT    2048    // C*D = GEMM K
#define TILE_M  128     // edges per block
#define THREADS 256     // 8 wave32s
#define LROW    132     // padded LDS row stride (floats): 132%64==4 -> bank rotation

union AFrag { v16h v; _Float16 h[16]; };
union BFrag { v16h v; float4 f4[2]; };

// ---------------------------------------------------------------------------
// Prep: pack fc_w (f32 [K=2048][N=128], row major) into f16 fragment order.
// Fragment group index = (kc*8 + t)*32 + lane, 16 contiguous halves per lane.
// B 16-bit 32x16 layout: lanes 0-15 hold k = kc*32 + {0..7,16..23} (n = t*16 + lane),
//                        lanes 16-31 hold k = kc*32 + {8..15,24..31}.
// ---------------------------------------------------------------------------
__global__ void pack_fcw_kernel(const float* __restrict__ fc_w,
                                _Float16* __restrict__ fc_wp) {
    int idx = blockIdx.x * blockDim.x + threadIdx.x;   // 64*8*32 = 16384 groups
    if (idx >= 64 * 8 * 32) return;
    int lane = idx & 31;
    int t    = (idx >> 5) & 7;
    int kc   = idx >> 8;
    int n    = t * 16 + (lane & 15);
    int kb   = kc * 32 + ((lane >> 4) << 3);
    _Float16* dst = fc_wp + (size_t)idx * 16;
#pragma unroll
    for (int hh = 0; hh < 16; ++hh)
        dst[hh] = (_Float16)fc_w[(size_t)(kb + hh + ((hh >> 3) << 3)) * Dm + n];
}

// ---------------------------------------------------------------------------
// Fused: gather + bn1 + conv-expand + relu (A on the fly) @ fc_w (WMMA f16)
//        + fc_b + bn3 + relu + dot(c_j) reduction.
// B fragments are register double-buffered one K-step ahead so the 16
// global_load_b128 per step are in flight during the previous step's WMMAs.
// ---------------------------------------------------------------------------
__global__ void __launch_bounds__(THREADS)
conve_fused_kernel(const float* __restrict__ h,  const float* __restrict__ g,
                   const int* __restrict__ edge_idx, const int* __restrict__ edge_type,
                   const float* __restrict__ conv_w, const float* __restrict__ conv_b,
                   const float* __restrict__ fc_b,
                   const float* __restrict__ bn1_g, const float* __restrict__ bn1_b,
                   const float* __restrict__ bn1_m, const float* __restrict__ bn1_v,
                   const float* __restrict__ bn3_g, const float* __restrict__ bn3_b,
                   const float* __restrict__ bn3_m, const float* __restrict__ bn3_v,
                   const _Float16* __restrict__ fc_wp,
                   float* __restrict__ out, int E) {
    __shared__ float sX0[TILE_M * LROW];   // x0 rows; reused for c_j in epilogue
    __shared__ float sX1[TILE_M * LROW];   // x1 rows
    __shared__ float sS3[Dm], sT3[Dm];
    __shared__ float sAl[Cc], sBe[Cc], sGa[Cc];

    const int tid   = threadIdx.x;
    const int eBase = blockIdx.x * TILE_M;

    // Fold bn1 into per-channel (alpha, beta, gamma); fold fc_b + bn3 into (s3, t3).
    if (tid < Cc) {
        float s1  = bn1_g[0] * rsqrtf(bn1_v[0] + EPS);
        float off = bn1_b[0] - bn1_m[0] * s1;
        float w0 = conv_w[2 * tid], w1 = conv_w[2 * tid + 1];
        sAl[tid] = w0 * s1;
        sBe[tid] = w1 * s1;
        sGa[tid] = conv_b[tid] + (w0 + w1) * off;
    }
    if (tid < Dm) {
        float s3 = bn3_g[tid] * rsqrtf(bn3_v[tid] + EPS);
        sS3[tid] = s3;
        sT3[tid] = (fc_b[tid] - bn3_m[tid]) * s3 + bn3_b[tid];
    }

    // Stage x0 = h[row], x1 = g[type] tiles in LDS (two threads per edge row).
    {
        int el   = tid >> 1;
        int half = tid & 1;
        int e    = eBase + el;
        int r0   = edge_idx[e];
        int rt   = edge_type[e];
        const float4* s0 = (const float4*)(h + (size_t)r0 * Dm + half * 64);
        const float4* s1 = (const float4*)(g + (size_t)rt * Dm + half * 64);
        float4* t0 = (float4*)(sX0 + el * LROW + half * 64);
        float4* t1 = (float4*)(sX1 + el * LROW + half * 64);
#pragma unroll
        for (int i = 0; i < 16; ++i) { t0[i] = s0[i]; t1[i] = s1[i]; }
    }
    __syncthreads();

    const int lane = tid & 31;
    const int wv   = tid >> 5;       // wave id: rows 16*wv .. 16*wv+15
    const int lhi  = lane >> 4;
    const int lm   = lane & 15;
    const int mrow = wv * 16 + lm;   // A-matrix row held by this lane

    const float* x0row = sX0 + mrow * LROW;
    const float* x1row = sX1 + mrow * LROW;

    v8f acc[8] = {};

    // Build A fragment for K-chunk kc (16-bit A 16x32 layout):
    // halves 0..7  -> j = jb + lhi*8 + 0..7 ;  halves 8..15 -> j = jb+16+lhi*8+0..7
    auto buildA = [&](AFrag& a, int kc) {
        const int c  = kc >> 2;
        const int jb = (kc & 3) * 32;
        const float al = sAl[c], be = sBe[c], ga = sGa[c];
        float x0v[16], x1v[16];
        const int j0 = jb + lhi * 8;
        const int j1 = jb + 16 + lhi * 8;
        *(float4*)&x0v[0]  = *(const float4*)(x0row + j0);
        *(float4*)&x0v[4]  = *(const float4*)(x0row + j0 + 4);
        *(float4*)&x0v[8]  = *(const float4*)(x0row + j1);
        *(float4*)&x0v[12] = *(const float4*)(x0row + j1 + 4);
        *(float4*)&x1v[0]  = *(const float4*)(x1row + j0);
        *(float4*)&x1v[4]  = *(const float4*)(x1row + j0 + 4);
        *(float4*)&x1v[8]  = *(const float4*)(x1row + j1);
        *(float4*)&x1v[12] = *(const float4*)(x1row + j1 + 4);
#pragma unroll
        for (int hh = 0; hh < 16; ++hh)
            a.h[hh] = (_Float16)fmaxf(al * x0v[hh] + be * x1v[hh] + ga, 0.0f);
    };

    auto loadB = [&](BFrag (&dst)[8], int kc) {
        const _Float16* bb = fc_wp + ((size_t)(kc * 8) * 32 + lane) * 16;
#pragma unroll
        for (int t = 0; t < 8; ++t) {
            const float4* bp = (const float4*)(bb + (size_t)t * 512);
            dst[t].f4[0] = bp[0];
            dst[t].f4[1] = bp[1];
        }
    };

    BFrag b0[8], b1[8];
    AFrag a0, a1;

    loadB(b0, 0);
    for (int kc = 0; kc < KTOT / 32; kc += 2) {
        // --- step kc: compute with b0, fetch b1 for kc+1 ---
        buildA(a0, kc);
        loadB(b1, kc + 1);
        if (kc + 2 < KTOT / 32)
            __builtin_prefetch(fc_wp + ((size_t)(kc + 2) * 256 + lane * 8) * 16, 0, 2);
#pragma unroll
        for (int t = 0; t < 8; ++t)
            acc[t] = __builtin_amdgcn_wmma_f32_16x16x32_f16(
                false, a0.v, false, b0[t].v, (short)0, acc[t], false, false);

        // --- step kc+1: compute with b1, fetch b0 for kc+2 ---
        buildA(a1, kc + 1);
        if (kc + 2 < KTOT / 32)
            loadB(b0, kc + 2);
#pragma unroll
        for (int t = 0; t < 8; ++t)
            acc[t] = __builtin_amdgcn_wmma_f32_16x16x32_f16(
                false, a1.v, false, b1[t].v, (short)0, acc[t], false, false);
    }

    // Epilogue: gather c_j = h[col] into sX0 (reused), then bn3+relu+dot.
    __syncthreads();
    {
        int el   = tid >> 1;
        int half = tid & 1;
        int e    = eBase + el;
        int ci   = edge_idx[E + e];
        const float4* sc = (const float4*)(h + (size_t)ci * Dm + half * 64);
        float4* tc = (float4*)(sX0 + el * LROW + half * 64);
#pragma unroll
        for (int i = 0; i < 16; ++i) tc[i] = sc[i];
    }
    __syncthreads();

    // C/D layout: lane (lm) = column n%16, lhi selects M 0-7 vs 8-15, reg r = M%8.
#pragma unroll
    for (int r = 0; r < 8; ++r) {
        const int erow = wv * 16 + lhi * 8 + r;      // edge row within block
        const float* cjrow = sX0 + erow * LROW;
        float partial = 0.0f;
#pragma unroll
        for (int t = 0; t < 8; ++t) {
            const int n = t * 16 + lm;
            float z = fmaxf(acc[t][r] * sS3[n] + sT3[n], 0.0f);
            partial += z * cjrow[n];
        }
        // Reduce across the 16-lane half-wave (wave32: masks stay in-group).
        partial += __shfl_xor(partial, 1, 32);
        partial += __shfl_xor(partial, 2, 32);
        partial += __shfl_xor(partial, 4, 32);
        partial += __shfl_xor(partial, 8, 32);
        if (lm == 0) out[eBase + erow] = partial;
    }
}

// ---------------------------------------------------------------------------
extern "C" void kernel_launch(void* const* d_in, const int* in_sizes, int n_in,
                              void* d_out, int out_size, void* d_ws, size_t ws_size,
                              hipStream_t stream) {
    const float* h         = (const float*)d_in[0];
    const float* g         = (const float*)d_in[1];
    const int*   edge_idx  = (const int*)d_in[2];
    const int*   edge_type = (const int*)d_in[3];
    const float* conv_w    = (const float*)d_in[4];
    const float* conv_b    = (const float*)d_in[5];
    const float* fc_w      = (const float*)d_in[6];
    const float* fc_b      = (const float*)d_in[7];
    const float* bn1_g     = (const float*)d_in[8];
    const float* bn1_b     = (const float*)d_in[9];
    const float* bn1_m     = (const float*)d_in[10];
    const float* bn1_v     = (const float*)d_in[11];
    const float* bn3_g     = (const float*)d_in[12];
    const float* bn3_b     = (const float*)d_in[13];
    const float* bn3_m     = (const float*)d_in[14];
    const float* bn3_v     = (const float*)d_in[15];
    float* out = (float*)d_out;
    const int E = in_sizes[3];

    _Float16* fc_wp = (_Float16*)d_ws;   // 2048*128*2 = 512 KB packed f16 B

    pack_fcw_kernel<<<64, 256, 0, stream>>>(fc_w, fc_wp);
    conve_fused_kernel<<<E / TILE_M, THREADS, 0, stream>>>(
        h, g, edge_idx, edge_type, conv_w, conv_b, fc_b,
        bn1_g, bn1_b, bn1_m, bn1_v, bn3_g, bn3_b, bn3_m, bn3_v,
        fc_wp, out, E);
}
